// LocalResnetPointnet_13778255086348
// MI455X (gfx1250) — matrix operand order repaired and checked by reading, not compiled, not run on hardware
//
#include <hip/hip_runtime.h>
#include <stdint.h>

typedef __bf16 v16bf __attribute__((ext_vector_type(16)));
typedef float  v8f   __attribute__((ext_vector_type(8)));
typedef unsigned int v4u __attribute__((ext_vector_type(4)));

#define HIDDEN 128
#define RESO 128
#define NSEG (RESO * RESO)

union Frag {
  v4u q[2];
  unsigned int u[8];
  v16bf v;
};

__device__ __forceinline__ unsigned short f32_to_bf16(float f) {
  __bf16 h = (__bf16)f;
  return __builtin_bit_cast(unsigned short, h);
}
// monotone float -> u32 encoding so we can max-pool with integer atomics
__device__ __forceinline__ unsigned enc_ord(float f) {
  unsigned u = __builtin_bit_cast(unsigned, f);
  return (u & 0x80000000u) ? ~u : (u | 0x80000000u);
}
__device__ __forceinline__ float dec_ord(unsigned u) {
  unsigned v = (u & 0x80000000u) ? (u & 0x7fffffffu) : ~u;
  return __builtin_bit_cast(float, v);
}

// B fragment (32x16 bf16, K x N): lane = N (mod 16), lanes 0-15 hold K+{0..7,16..23}, lanes 16-31 +8;
// weights stored transposed [N][K] so the 16 K-values per lane are contiguous.
__device__ __forceinline__ Frag load_bfrag(const unsigned short* __restrict__ wT,
                                           int Kdim, int nt, int kt, int lane) {
  int n  = nt * 16 + (lane & 15);
  int kb = kt * 32 + ((lane >> 4) ? 8 : 0);
  const unsigned short* p = wT + (size_t)n * Kdim + kb;
  Frag f;
  f.q[0] = *(const v4u*)(p);
  f.q[1] = *(const v4u*)(p + 16);
  return f;
}

// A fragment (16x32 bf16): lane l holds row m=l&15; lanes 0-15: K = {kt*32+0..7, +16..23};
// lanes 16-31: +8. Row is contiguous bf16 in LDS -> two aligned b128 loads.
__device__ __forceinline__ Frag gather_afrag_lds(const unsigned short* row, int kt, int hi) {
  const unsigned short* p = row + kt * 32 + hi * 8;
  Frag f;
  f.q[0] = *(const v4u*)(p);
  f.q[1] = *(const v4u*)(p + 16);
  return f;
}

__device__ __forceinline__ Frag relu_frag(Frag a) {
  Frag r;
#pragma unroll
  for (int i = 0; i < 8; ++i) {
    unsigned u = a.u[i];
    unsigned msk = ((u & 0x80008000u) >> 15) * 0xFFFFu; // 0xFFFF per negative bf16 half
    r.u[i] = u & ~msk;
  }
  return r;
}

// One ResnetBlockFC over 64 points per workgroup (4 waves x 16 points).
// x (16x256 bf16) staged in LDS; net = relu(x)@w0+b0; h = relu(net)@w1+b1 + x@ws.
__global__ __launch_bounds__(128) void resblock_kernel(
    const float* __restrict__ pts,
    const float* __restrict__ fcw, const float* __restrict__ fcb,
    const unsigned short* __restrict__ w0T,
    const unsigned short* __restrict__ w1T,
    const unsigned short* __restrict__ wsT,
    const float* __restrict__ b0, const float* __restrict__ b1,
    const float* hin, float* hout,
    const float* __restrict__ pooledIn,
    unsigned* __restrict__ partials,
    int T, int isFirst, int doPool) {
  __shared__ __align__(16) unsigned short xtile[4][16][256];   // 32 KB bf16 x
  __shared__ __align__(16) unsigned short nettile[4][16][128]; // 16 KB bf16 relu(net)
  __shared__ unsigned pooledEnc[128];

  const int tid  = threadIdx.x;
  const int w    = tid >> 5;
  const int lane = tid & 31;
  const int b    = blockIdx.y;
  const int t0   = blockIdx.x * 64 + w * 16;

  if (tid < 128) pooledEnc[tid] = 0u;

  if (isFirst) {
    // x = p @ fc_pos_w + fc_pos_b (K=3, done scalar)
    for (int r = 0; r < 16; ++r) {
      const float* pr = pts + ((size_t)b * T + (t0 + r)) * 3;
      float p0 = pr[0], p1 = pr[1], p2 = pr[2];
#pragma unroll
      for (int j = 0; j < 8; ++j) {
        int c = j * 32 + lane;
        float v = fcb[c] + p0 * fcw[c] + p1 * fcw[256 + c] + p2 * fcw[512 + c];
        xtile[w][r][c] = f32_to_bf16(v);
      }
    }
  } else {
    // x = concat(hfeat, broadcast(pooled))
    for (int r = 0; r < 16; ++r) {
      const float* hr = hin + ((size_t)b * T + (t0 + r)) * HIDDEN;
#pragma unroll
      for (int j = 0; j < 8; ++j) {
        int c = j * 32 + lane;
        float v = (c < HIDDEN) ? hr[c] : pooledIn[b * HIDDEN + (c - HIDDEN)];
        xtile[w][r][c] = f32_to_bf16(v);
      }
    }
  }
  __syncthreads();

  const int m  = lane & 15;
  const int hi = lane >> 4;
  const unsigned short* xrow = &xtile[w][m][0];

  // ---- phase 1: net = relu(x) @ w0 + b0 ; keep relu(net) bf16 in LDS ----
  {
    Frag arelu[8];
#pragma unroll
    for (int kt = 0; kt < 8; ++kt)
      arelu[kt] = relu_frag(gather_afrag_lds(xrow, kt, hi));

#pragma unroll
    for (int nt = 0; nt < 8; ++nt) {
      float bias = b0[nt * 16 + m]; // C/D: lane holds one column -> bias splat
      v8f acc;
#pragma unroll
      for (int i = 0; i < 8; ++i) acc[i] = bias;
#pragma unroll
      for (int kt = 0; kt < 8; ++kt) {
        Frag bf = load_bfrag(w0T, 256, nt, kt, lane);
        acc = __builtin_amdgcn_wmma_f32_16x16x32_bf16(false, arelu[kt].v, false, bf.v,
                                                      (short)0, acc, false, false);
      }
#pragma unroll
      for (int i = 0; i < 8; ++i) {
        int m2 = i + hi * 8; // C/D row mapping
        float rv = acc[i] > 0.0f ? acc[i] : 0.0f;
        nettile[w][m2][nt * 16 + m] = f32_to_bf16(rv);
      }
    }
  }

  // ---- phase 2: h = relu(net) @ w1 + b1 + x @ ws ----
  {
    Frag anet[4];
    const unsigned short* nrow = &nettile[w][m][0];
#pragma unroll
    for (int kt = 0; kt < 4; ++kt) anet[kt] = gather_afrag_lds(nrow, kt, hi);
    Frag araw[8];
#pragma unroll
    for (int kt = 0; kt < 8; ++kt) araw[kt] = gather_afrag_lds(xrow, kt, hi);

#pragma unroll
    for (int nt = 0; nt < 8; ++nt) {
      float bias = b1[nt * 16 + m];
      v8f acc;
#pragma unroll
      for (int i = 0; i < 8; ++i) acc[i] = bias;
#pragma unroll
      for (int kt = 0; kt < 4; ++kt) {
        Frag bf = load_bfrag(w1T, 128, nt, kt, lane);
        acc = __builtin_amdgcn_wmma_f32_16x16x32_bf16(false, anet[kt].v, false, bf.v,
                                                      (short)0, acc, false, false);
      }
#pragma unroll
      for (int kt = 0; kt < 8; ++kt) {
        Frag bf = load_bfrag(wsT, 256, nt, kt, lane);
        acc = __builtin_amdgcn_wmma_f32_16x16x32_bf16(false, araw[kt].v, false, bf.v,
                                                      (short)0, acc, false, false);
      }
      int cg = nt * 16 + m;
      float lmax = -3.4e38f;
#pragma unroll
      for (int i = 0; i < 8; ++i) {
        int m2 = i + hi * 8;
        hout[((size_t)b * T + (t0 + m2)) * HIDDEN + cg] = acc[i];
        lmax = acc[i] > lmax ? acc[i] : lmax;
      }
      if (doPool) atomicMax(&pooledEnc[cg], enc_ord(lmax));
    }
  }

  __syncthreads();
  if (doPool && tid < 128)
    partials[((size_t)b * gridDim.x + blockIdx.x) * 128 + tid] = pooledEnc[tid];
}

__global__ void reduce_pool_kernel(const unsigned* __restrict__ partials,
                                   float* __restrict__ pooled, int nwg) {
  int b = blockIdx.x, c = threadIdx.x;
  unsigned mx = 0u;
  for (int wg = 0; wg < nwg; ++wg) {
    unsigned v = partials[((size_t)b * nwg + wg) * 128 + c];
    mx = v > mx ? v : mx;
  }
  pooled[b * 128 + c] = dec_ord(mx);
}

__global__ void convert_weights_kernel(const float* __restrict__ w0,
                                       const float* __restrict__ w1,
                                       const float* __restrict__ ws,
                                       unsigned short* __restrict__ w0T,
                                       unsigned short* __restrict__ w1T,
                                       unsigned short* __restrict__ wsT) {
  int i = blockIdx.x * blockDim.x + threadIdx.x;
  const int n0 = 5 * 128 * 256;
  const int n1 = 5 * 128 * 128;
  if (i < n0) {
    int blk = i / (128 * 256), r = i % (128 * 256);
    int n = r / 256, k = r % 256;
    w0T[i] = f32_to_bf16(w0[(size_t)blk * 256 * 128 + (size_t)k * 128 + n]);
  } else if (i < n0 + n1) {
    int j = i - n0;
    int blk = j / (128 * 128), r = j % (128 * 128);
    int n = r / 128, k = r % 128;
    w1T[j] = f32_to_bf16(w1[(size_t)blk * 128 * 128 + (size_t)k * 128 + n]);
  } else if (i < n0 + n1 + n0) {
    int j = i - n0 - n1;
    int blk = j / (128 * 256), r = j % (128 * 256);
    int n = r / 256, k = r % 256;
    wsT[j] = f32_to_bf16(ws[(size_t)blk * 256 * 128 + (size_t)k * 128 + n]);
  }
}

__global__ __launch_bounds__(128) void scatter_kernel(const float* __restrict__ pts,
                                                      const float* __restrict__ hbuf,
                                                      float* out, float* counts, int T) {
  int t = blockIdx.x, b = blockIdx.y, c = threadIdx.x;
  const float* pr = pts + ((size_t)b * T + t) * 3;
  float x0 = pr[0] / 1.101f + 0.5f;
  float x2 = pr[2] / 1.101f + 0.5f;
  x0 = fminf(fmaxf(x0, 0.0f), 1.0f - 1e-6f);
  x2 = fminf(fmaxf(x2, 0.0f), 1.0f - 1e-6f);
  int xi0 = (int)(x0 * (float)RESO); xi0 = xi0 > RESO - 1 ? RESO - 1 : xi0;
  int xi1 = (int)(x2 * (float)RESO); xi1 = xi1 > RESO - 1 ? RESO - 1 : xi1;
  int idx = xi0 + RESO * xi1;
  float v = hbuf[((size_t)b * T + t) * HIDDEN + c];
  atomicAdd(&out[((size_t)(b * HIDDEN + c)) * NSEG + idx], v);
  if (c == 0) atomicAdd(&counts[b * NSEG + idx], 1.0f);
}

__global__ void normalize_kernel(float* out, const float* __restrict__ counts, int total) {
  int i = blockIdx.x * blockDim.x + threadIdx.x;
  if (i >= total) return;
  int b = i >> 21; // 128 channels * 16384 bins = 2^21 per batch
  int s = i & (NSEG - 1);
  float c = counts[b * NSEG + s];
  out[i] = out[i] / fmaxf(c, 1.0f);
}

extern "C" void kernel_launch(void* const* d_in, const int* in_sizes, int n_in,
                              void* d_out, int out_size, void* d_ws, size_t ws_size,
                              hipStream_t stream) {
  const float* pts = (const float*)d_in[0];
  const float* fcw = (const float*)d_in[1];
  const float* fcb = (const float*)d_in[2];
  const float* w0  = (const float*)d_in[3];
  const float* b0  = (const float*)d_in[4];
  const float* w1  = (const float*)d_in[5];
  const float* b1  = (const float*)d_in[6];
  const float* ws  = (const float*)d_in[7];

  const int B = 2;
  const int T = in_sizes[0] / (B * 3); // p is (B,T,3)
  const int nwg = T / 64;

  char* base = (char*)d_ws;
  size_t o = 0;
  float* hbuf = (float*)(base + o);                   o += (size_t)B * T * HIDDEN * 4;
  unsigned short* w0T = (unsigned short*)(base + o);  o += (size_t)5 * 128 * 256 * 2;
  unsigned short* w1T = (unsigned short*)(base + o);  o += (size_t)5 * 128 * 128 * 2;
  unsigned short* wsT = (unsigned short*)(base + o);  o += (size_t)5 * 128 * 256 * 2;
  unsigned* partials = (unsigned*)(base + o);         o += (size_t)B * nwg * 128 * 4;
  float* pooled = (float*)(base + o);                 o += (size_t)B * 128 * 4;
  float* counts = (float*)(base + o);                 o += (size_t)B * NSEG * 4;

  hipMemsetAsync(d_out, 0, (size_t)out_size * sizeof(float), stream);
  hipMemsetAsync(counts, 0, (size_t)B * NSEG * sizeof(float), stream);

  int cvt_total = 5 * 128 * 256 + 5 * 128 * 128 + 5 * 128 * 256;
  convert_weights_kernel<<<(cvt_total + 255) / 256, 256, 0, stream>>>(w0, w1, ws, w0T, w1T, wsT);

  for (int i = 0; i < 5; ++i) {
    resblock_kernel<<<dim3(nwg, B), 128, 0, stream>>>(
        pts, fcw, fcb,
        w0T + (size_t)i * 128 * 256,
        w1T + (size_t)i * 128 * 128,
        wsT + (size_t)i * 128 * 256,
        b0 + i * 128, b1 + i * 128,
        hbuf, hbuf, pooled, partials,
        T, i == 0 ? 1 : 0, i < 4 ? 1 : 0);
    if (i < 4)
      reduce_pool_kernel<<<B, 128, 0, stream>>>(partials, pooled, nwg);
  }

  scatter_kernel<<<dim3(T, B), 128, 0, stream>>>(pts, hbuf, (float*)d_out, counts, T);

  int total = B * HIDDEN * NSEG;
  normalize_kernel<<<(total + 255) / 256, 256, 0, stream>>>((float*)d_out, counts, total);
}
